// WindowAttention_4587025072234
// MI455X (gfx1250) — matrix operand order repaired and checked by reading, not compiled, not run on hardware
//
#include <hip/hip_runtime.h>
#include <cstddef>

// ---------------------------------------------------------------------------
// Window attention for MI455X (gfx1250, wave32, WMMA bf16 16x16x32 + TDM).
// Pipeline: f32->bf16 convert, q/kv GEMMs (WMMA, TDM-staged double-buffered
// LDS), fused attention block (scores -> bias -> mask -> softmax -> p@v in
// LDS, q/k tiles staged by TDM), output projection.
// ---------------------------------------------------------------------------

typedef __attribute__((ext_vector_type(16))) __bf16 v16bf;
typedef __attribute__((ext_vector_type(8)))  float  v8f;
typedef __attribute__((ext_vector_type(4)))  unsigned int v4u;
typedef __attribute__((ext_vector_type(8)))  int v8i;
typedef unsigned short u16_t;

#define DIMC 512
#define HC   8
#define HDC  64
#define NTOK 65    // 1 + W
#define MTOK 129   // 1 + R
#define BWIN 512   // B_ windows total
#define NWIN 64    // nW

__device__ __forceinline__ u16_t f2bf(float f) {
  unsigned u = __float_as_uint(f);
  u += 0x7FFFu + ((u >> 16) & 1u);       // round-to-nearest-even
  return (u16_t)(u >> 16);
}

union U16V { uint4 u[2]; v16bf v; };

// A operand (16x32 MxK, bf16): lane L<16 holds row L, K chunks {ks..+7, ks+16..+23};
// lane L>=16 holds row L-16, K chunks {ks+8..+15, ks+24..+31}.
__device__ __forceinline__ v16bf ld_a(const u16_t* rowp, int ks, int sel) {
  U16V t;
  t.u[0] = *(const uint4*)(rowp + ks + sel * 8);
  t.u[1] = *(const uint4*)(rowp + ks + 16 + sel * 8);
  return t.v;
}
// B operand (32x16 KxN, bf16): lane holds one column (= one row of the NxK
// weight matrix); half-wave sel holds K chunks {ks+sel*8..+7, ks+16+sel*8..+7}.
__device__ __forceinline__ v16bf ld_b(const u16_t* rowp, int ks, int sel) {
  U16V t;
  const u16_t* p = rowp + ks + sel * 8;
  t.u[0] = *(const uint4*)(p);
  t.u[1] = *(const uint4*)(p + 16);
  return t.v;
}
__device__ __forceinline__ v16bf zero16() {
  U16V t; uint4 z = {0u,0u,0u,0u}; t.u[0] = z; t.u[1] = z; return t.v;
}
__device__ __forceinline__ v8f wmma_bf16(v16bf a, v16bf b, v8f c) {
  return __builtin_amdgcn_wmma_f32_16x16x32_bf16(false, a, false, b,
                                                 (short)0, c, false, false);
}

// ---------------------------------------------------------------------------
// Tensor Data Mover: 2D tile (tile_w x tile_h elements, bf16) from a
// row-major tensor (row stride = row_stride elements) into contiguous LDS
// rows of tile_w elements. D# groups per CDNA5 ISA.
// Inline asm (2-group form, VADDR2/3 = NULL) for toolchain portability.
// ---------------------------------------------------------------------------
__device__ __forceinline__ void tdm_load_2d(const void* gsrc, void* lds_dst,
                                            unsigned tile_w, unsigned tile_h,
                                            unsigned tensor_w, unsigned tensor_h,
                                            unsigned row_stride) {
  unsigned long long ga = (unsigned long long)(size_t)gsrc;
  unsigned la = (unsigned)(size_t)lds_dst;   // flat addr[31:0] == LDS offset
  v4u g0;
  g0[0] = 1u;                                       // count=1, user descriptor
  g0[1] = la;                                       // lds_addr
  g0[2] = (unsigned)ga;                             // global_addr[31:0]
  g0[3] = (unsigned)((ga >> 32) & 0x01FFFFFFu) |    // global_addr[56:32]
          0x80000000u;                              // type=2 ("image")
  v8i g1;
  g1[0] = (int)(1u << 16);                          // data_size=1 (2 bytes)
  g1[1] = (int)((tensor_w & 0xFFFFu) << 16);        // tensor_dim0[15:0]
  g1[2] = (int)(((tensor_w >> 16) & 0xFFFFu) |      // tensor_dim0[31:16]
                ((tensor_h & 0xFFFFu) << 16));      // tensor_dim1[15:0]
  g1[3] = (int)(((tensor_h >> 16) & 0xFFFFu) |      // tensor_dim1[31:16]
                (tile_w << 16));                    // tile_dim0
  g1[4] = (int)(tile_h & 0xFFFFu);                  // tile_dim1 (tile_dim2=0)
  g1[5] = (int)row_stride;                          // tensor_dim0_stride[31:0]
  g1[6] = 0;                                        // stride[47:32], dim1_stride lo
  g1[7] = 0;
  asm volatile("tensor_load_to_lds %0, %1" :: "s"(g0), "s"(g1) : "memory");
}

__device__ __forceinline__ void wait_tensorcnt0() {
#if __has_builtin(__builtin_amdgcn_s_wait_tensorcnt)
  __builtin_amdgcn_s_wait_tensorcnt(0);
#else
  asm volatile("s_wait_tensorcnt 0x0" ::: "memory");
#endif
}

// ---------------------------------------------------------------------------
// f32 -> bf16 conversion
// ---------------------------------------------------------------------------
__global__ void cvt_f32_bf16(const float* __restrict__ src,
                             u16_t* __restrict__ dst, int n) {
  int i = blockIdx.x * 256 + threadIdx.x;
  if (i < n) dst[i] = f2bf(src[i]);
}

// ---------------------------------------------------------------------------
// C(MxN) = A(MxK,bf16) * B(NxK,bf16)^T + bias ; optional scale; out bf16/f32.
// M % 128 == 0, N % 64 == 0, K % 64 == 0. Block = 256 threads = 8 waves,
// block tile 128x64, wave tile 32x32. LDS staging by TDM, double-buffered:
// wave 0 issues next-stage descriptors, everyone computes current stage.
// ---------------------------------------------------------------------------
__global__ __launch_bounds__(256, 2)
void gemm_bf16_kernel(const u16_t* __restrict__ A, const u16_t* __restrict__ B,
                      const float* __restrict__ bias, float scale,
                      u16_t* __restrict__ outB, float* __restrict__ outF,
                      int M, int N, int K) {
  __shared__ __align__(16) u16_t As[2][128 * 64];
  __shared__ __align__(16) u16_t Bs[2][64 * 64];

  const int tid = threadIdx.x;
  const int lane = tid & 31, wave = tid >> 5;
  const int row16 = lane & 15, sel = lane >> 4;
  const int m0 = blockIdx.x * 128, n0 = blockIdx.y * 64;
  const int wm = wave & 3, wn = wave >> 2;

  const u16_t* Abase = A + (size_t)m0 * K;
  const u16_t* Bbase = B + (size_t)n0 * K;

  v8f c00 = {}, c01 = {}, c10 = {}, c11 = {};

  // prologue: DMA stage 0 into buffer 0
  if (wave == 0) {
    tdm_load_2d(Abase, As[0], 64, 128, (unsigned)K, (unsigned)M, (unsigned)K);
    tdm_load_2d(Bbase, Bs[0], 64, 64,  (unsigned)K, (unsigned)N, (unsigned)K);
    wait_tensorcnt0();
  }
  __syncthreads();

  const int nstage = K >> 6;
  for (int s = 0; s < nstage; ++s) {
    const int cur = s & 1, nxt = cur ^ 1;
    const bool more = (s + 1) < nstage;
    if (wave == 0 && more) {                       // async prefetch next stage
      tdm_load_2d(Abase + (s + 1) * 64, As[nxt], 64, 128,
                  (unsigned)K, (unsigned)M, (unsigned)K);
      tdm_load_2d(Bbase + (s + 1) * 64, Bs[nxt], 64, 64,
                  (unsigned)K, (unsigned)N, (unsigned)K);
    }
    for (int ks = 0; ks < 64; ks += 32) {
      v16bf a0 = ld_a(As[cur] + (wm * 32 + row16) * 64, ks, sel);
      v16bf a1 = ld_a(As[cur] + (wm * 32 + 16 + row16) * 64, ks, sel);
      v16bf b0 = ld_b(Bs[cur] + (wn * 32 + row16) * 64, ks, sel);
      v16bf b1 = ld_b(Bs[cur] + (wn * 32 + 16 + row16) * 64, ks, sel);
      c00 = wmma_bf16(a0, b0, c00);
      c01 = wmma_bf16(a0, b1, c01);
      c10 = wmma_bf16(a1, b0, c10);
      c11 = wmma_bf16(a1, b1, c11);
    }
    if (wave == 0 && more) wait_tensorcnt0();      // next buffer ready
    __syncthreads();
  }

  // epilogue: C layout = VGPR j -> M = j + sel*8, N = lane%16
  const int rowBase = m0 + wm * 32 + sel * 8;
  const int colBase = n0 + wn * 32 + row16;
  v8f acc[4] = {c00, c01, c10, c11};
  for (int q = 0; q < 4; ++q) {
    int row = rowBase + ((q >> 1) << 4);
    int col = colBase + ((q & 1) << 4);
    float bv = bias[col];
    v8f c = acc[q];
    for (int j = 0; j < 8; ++j) {
      float v = (c[j] + bv) * scale;
      size_t idx = (size_t)(row + j) * N + col;
      if (outF) outF[idx] = v;
      else      outB[idx] = f2bf(v);
    }
  }
}

// ---------------------------------------------------------------------------
// Fused attention: one workgroup per (window b_, head h).
// LDS (62,992 B): region A = q(80x64 bf16) + k(144x64 bf16), reused for
// vT(64x160 bf16); region S = scores 65x132 f32, converted in-place to
// bf16 probabilities padded to K=160. q/k tiles staged by TDM.
// ---------------------------------------------------------------------------
__global__ __launch_bounds__(256, 1)
void attn_kernel(const u16_t* __restrict__ qb, const u16_t* __restrict__ kvb,
                 const float* __restrict__ rel_table,
                 const float* __restrict__ cls_self,
                 const float* __restrict__ cls_up,
                 const float* __restrict__ cls_down,
                 const int* __restrict__ mask_left,
                 const int* __restrict__ mask_right,
                 u16_t* __restrict__ aob) {
  __shared__ __align__(16) unsigned char smem[62992];
  u16_t* q_lds  = (u16_t*)smem;                  // 80  x 64
  u16_t* k_lds  = (u16_t*)(smem + 10240);        // 144 x 64
  u16_t* vT_lds = (u16_t*)smem;                  // 64  x 160 (aliases q/k)
  float* s_lds  = (float*)(smem + 28672);        // 65  x 132

  const int bh = blockIdx.x;
  const int h = bh & 7, b = bh >> 3;
  const int w = b & (NWIN - 1);
  const int tid = threadIdx.x;
  const int lane = tid & 31, wave = tid >> 5;
  const int row16 = lane & 15, sel = lane >> 4;

  const u16_t* qg = qb  + (size_t)b * NTOK * DIMC + h * HDC;
  const u16_t* kg = kvb + (size_t)b * MTOK * (2 * DIMC) + h * HDC;
  const u16_t* vg = kg + DIMC;

  // --- TDM-stage q (65x64 @ stride 512) and k (129x64 @ stride 1024) ---
  if (wave == 0) {
    tdm_load_2d(qg, q_lds, 64, NTOK, DIMC, NTOK, DIMC);
    tdm_load_2d(kg, k_lds, 64, MTOK, 2 * DIMC, MTOK, 2 * DIMC);
  }
  // zero the padded tile rows (q rows 65..79, k rows 129..143) meanwhile
  uint4 z4 = {0u,0u,0u,0u};
  for (int idx = tid; idx < 120; idx += 256)
    *(uint4*)(q_lds + NTOK * 64 + idx * 8) = z4;
  for (int idx = tid; idx < 120; idx += 256)
    *(uint4*)(k_lds + MTOK * 64 + idx * 8) = z4;
  if (wave == 0) wait_tensorcnt0();
  __syncthreads();

  // --- scores = q (scaled) @ k^T, + bias, mask -> s_lds --- 5x9 tiles
  for (int t = wave; t < 45; t += 8) {
    int mt = t / 9, nt = t % 9;
    v8f c = {};
    for (int ks = 0; ks < HDC; ks += 32) {
      v16bf a = ld_a(q_lds + (mt * 16 + row16) * 64, ks, sel);
      v16bf bm = ld_b(k_lds + (nt * 16 + row16) * 64, ks, sel);
      c = wmma_bf16(a, bm, c);
    }
    int col = nt * 16 + row16;
    if (col < MTOK) {
      for (int j = 0; j < 8; ++j) {
        int row = mt * 16 + j + sel * 8;
        if (row < NTOK) {
          float sc = c[j];
          float bias;
          if (row == 0 && col == 0)      bias = cls_self[h];
          else if (row == 0)             bias = cls_up[h * 128 + (col - 1)];
          else if (col == 0)             bias = cls_down[h * 64 + (row - 1)];
          else                           bias = rel_table[(row - col + 127) * HC + h];
          sc += bias;
          int masked = 0;
          if (w < 4)         masked  = mask_left [(w * NTOK + row) * MTOK + col];
          if (w >= NWIN - 4) masked |= mask_right[((w - (NWIN - 4)) * NTOK + row) * MTOK + col];
          s_lds[row * 132 + col] = masked ? -3.402823466e+38f : sc;
        }
      }
    }
  }
  __syncthreads();

  // --- load v transposed into region A (64 x 160, pad m 129..159 = 0) ---
  for (int idx = tid; idx < 64 * 31; idx += 256) {
    int d = idx / 31, m = 129 + idx % 31;
    vT_lds[d * 160 + m] = 0;
  }
  for (int idx = tid; idx < MTOK * 8; idx += 256) {
    int m = idx >> 3, d8 = (idx & 7) << 3;
    union { uint4 u; u16_t s[8]; } t;
    t.u = *(const uint4*)(vg + (size_t)m * (2 * DIMC) + d8);
    for (int e = 0; e < 8; ++e) vT_lds[(d8 + e) * 160 + m] = t.s[e];
  }

  // --- row softmax (thread t owns row t), in-place f32 -> bf16 probs ---
  if (tid < NTOK) {
    float* srow = s_lds + tid * 132;
    float mx = -3.402823466e+38f;
    for (int j = 0; j < MTOK; ++j) mx = fmaxf(mx, srow[j]);
    float sum = 0.f;
    for (int j = 0; j < MTOK; ++j) sum += __expf(srow[j] - mx);
    float inv = 1.f / sum;
    u16_t* prow = (u16_t*)srow;
    for (int j = 0; j < MTOK; ++j) {
      float p = __expf(srow[j] - mx) * inv;   // read f32 before overwrite (2j+1 < 4j)
      prow[j] = f2bf(p);
    }
    for (int j = MTOK; j < 160; ++j) prow[j] = 0;
  }
  __syncthreads();

  // --- out = probs (65x160 bf16) @ v (160x64) --- 5x4 tiles, K=160
  for (int t = wave; t < 20; t += 8) {
    int mt = t >> 2, nt = t & 3;
    int arow = mt * 16 + row16;
    v8f c = {};
    for (int ks = 0; ks < 160; ks += 32) {
      v16bf a = (arow < NTOK)
                    ? ld_a((const u16_t*)(s_lds + arow * 132), ks, sel)
                    : zero16();
      v16bf bm = ld_b(vT_lds + (nt * 16 + row16) * 160, ks, sel);
      c = wmma_bf16(a, bm, c);
    }
    int col = h * HDC + nt * 16 + row16;
    for (int j = 0; j < 8; ++j) {
      int row = mt * 16 + j + sel * 8;
      if (row < NTOK)
        aob[(size_t)b * NTOK * DIMC + (size_t)row * DIMC + col] = f2bf(c[j]);
    }
  }
}

// ---------------------------------------------------------------------------
// Host-side launch
// ---------------------------------------------------------------------------
extern "C" void kernel_launch(void* const* d_in, const int* in_sizes, int n_in,
                              void* d_out, int out_size, void* d_ws, size_t ws_size,
                              hipStream_t stream) {
  (void)in_sizes; (void)n_in; (void)out_size; (void)ws_size;

  const float* x        = (const float*)d_in[0];   // (512, 65, 512)
  const float* x_       = (const float*)d_in[1];   // (512, 129, 512)
  const int*   mask_l   = (const int*)d_in[2];     // (4, 65, 129)
  const int*   mask_r   = (const int*)d_in[3];     // (4, 65, 129)
  /* d_in[4] = nW (64, hardcoded) */
  const float* rel_tab  = (const float*)d_in[5];   // (191, 8)
  const float* cls_self = (const float*)d_in[6];   // (8,)
  const float* cls_up   = (const float*)d_in[7];   // (8, 128)
  const float* cls_down = (const float*)d_in[8];   // (8, 64)
  const float* Wq       = (const float*)d_in[9];   // (512, 512)
  const float* bq       = (const float*)d_in[10];  // (512,)
  const float* Wkv      = (const float*)d_in[11];  // (1024, 512)
  const float* bkv      = (const float*)d_in[12];  // (1024,)
  const float* Wp       = (const float*)d_in[13];  // (512, 512)
  const float* bp       = (const float*)d_in[14];  // (512,)
  float* out = (float*)d_out;                      // (512, 65, 512)

  const size_t nX  = (size_t)BWIN * NTOK * DIMC;       // 17,039,360
  const size_t nX_ = (size_t)BWIN * MTOK * DIMC;       // 33,816,576
  const size_t nWq = (size_t)DIMC * DIMC;
  const size_t nWkv = (size_t)2 * DIMC * DIMC;
  const size_t nKV = (size_t)BWIN * MTOK * 2 * DIMC;   // 67,633,152

  unsigned char* ws = (unsigned char*)d_ws;
  size_t off = 0;
  u16_t* xb   = (u16_t*)(ws + off); off += nX  * 2;
  u16_t* x_b  = (u16_t*)(ws + off); off += nX_ * 2;
  u16_t* Wqb  = (u16_t*)(ws + off); off += nWq * 2;
  u16_t* Wkvb = (u16_t*)(ws + off); off += nWkv * 2;
  u16_t* Wpb  = (u16_t*)(ws + off); off += nWq * 2;
  u16_t* qb   = (u16_t*)(ws + off); off += nX  * 2;
  u16_t* kvb  = (u16_t*)(ws + off); off += nKV * 2;
  u16_t* aob  = (u16_t*)(ws + off); off += nX  * 2;

  // 1) f32 -> bf16 conversions
  cvt_f32_bf16<<<(int)((nX  + 255) / 256), 256, 0, stream>>>(x,   xb,   (int)nX);
  cvt_f32_bf16<<<(int)((nX_ + 255) / 256), 256, 0, stream>>>(x_,  x_b,  (int)nX_);
  cvt_f32_bf16<<<(int)((nWq + 255) / 256), 256, 0, stream>>>(Wq,  Wqb,  (int)nWq);
  cvt_f32_bf16<<<(int)((nWkv + 255) / 256), 256, 0, stream>>>(Wkv, Wkvb, (int)nWkv);
  cvt_f32_bf16<<<(int)((nWq + 255) / 256), 256, 0, stream>>>(Wp,  Wpb,  (int)nWq);

  // 2) q = (x @ Wq^T + bq) * HD^-0.5   -> bf16  (M=33280, N=512, K=512)
  gemm_bf16_kernel<<<dim3(33280 / 128, 512 / 64), 256, 0, stream>>>(
      xb, Wqb, bq, 0.125f, qb, nullptr, 33280, 512, 512);

  // 3) kv = x_ @ Wkv^T + bkv           -> bf16  (M=66048, N=1024, K=512)
  gemm_bf16_kernel<<<dim3(66048 / 128, 1024 / 64), 256, 0, stream>>>(
      x_b, Wkvb, bkv, 1.0f, kvb, nullptr, 66048, 1024, 512);

  // 4) fused attention per (window, head)
  attn_kernel<<<BWIN * HC, 256, 0, stream>>>(qb, kvb, rel_tab, cls_self,
                                             cls_up, cls_down, mask_l, mask_r,
                                             aob);

  // 5) out = attn_out @ Wp^T + bp      -> f32   (M=33280, N=512, K=512)
  gemm_bf16_kernel<<<dim3(33280 / 128, 512 / 64), 256, 0, stream>>>(
      aob, Wpb, bp, 1.0f, nullptr, out, 33280, 512, 512);
}
